// GPUEuclideanNeighborFinder_52596169506942
// MI455X (gfx1250) — compile-verified, matrix operand order checked by reading, not run on hardware
//
#include <hip/hip_runtime.h>

// ---------------------------------------------------------------------------
// GPUEuclideanNeighborFinder for MI455X (gfx1250, wave32, WMMA)
//   X: [8192, 512] f32, Y: [65536, 512] f32 -> indices of 8 nearest Y rows
//   Rank by  ||y||^2 - 2 * (X . Y^T)   (||x||^2 is a per-row constant).
//   GEMM via bf16x2 split on v_wmma_f32_16x16x32_bf16 (f32 accumulate),
//   ping-pong LDS B-tiles, one barrier per K-slab, unroll capped at 2 to
//   keep VGPR pressure (and thus occupancy) in a good range.
// ---------------------------------------------------------------------------

#define NX        8192
#define NY        65536
#define DDIM      512
#define KNEIGH    8
#define M_BLK     128            // X rows per workgroup
#define N_BLK     64             // Y rows (cols of sim) per tile step
#define KT        32             // K per WMMA slab
#define K_STEPS   (DDIM / KT)    // 16
#define N_CHUNKS  32             // Y split among this many col-chunks
#define CHUNK_COLS (NY / N_CHUNKS)        // 2048
#define NT_PER_CHUNK (CHUNK_COLS / N_BLK) // 32
#define BPAD      40             // padded LDS row stride (bf16 elems): 16B aligned, conflict-free
#define DPAD      65             // padded LDS row stride for dist tile (floats)

typedef __attribute__((ext_vector_type(16))) __bf16 v16bf;
typedef __attribute__((ext_vector_type(8)))  float  v8f;

union FragB16 { v16bf v; uint4 u[2]; };

// native bf16 split: x ~= hi + lo (both RNE via v_cvt hardware path)
__device__ __forceinline__ void split_bf16(float x, __bf16& h, __bf16& l) {
    h = (__bf16)x;
    l = (__bf16)(x - (float)h);
}

// ---------------------------------------------------------------------------
// Phase 0a: ||y||^2 per Y row (one wave per row, f32 exact)
// ---------------------------------------------------------------------------
__global__ __launch_bounds__(256) void ysq_kernel(const float* __restrict__ Y,
                                                  float* __restrict__ ysq) {
    const int lane = threadIdx.x & 31;
    const int w    = threadIdx.x >> 5;
    const int row  = blockIdx.x * 8 + w;          // grid = NY/8 blocks
    const float* yr = Y + (size_t)row * DDIM;
    float s = 0.f;
#pragma unroll
    for (int j = 0; j < DDIM / 32; ++j) {
        float v = yr[j * 32 + lane];
        s += v * v;
    }
#pragma unroll
    for (int m = 16; m >= 1; m >>= 1) s += __shfl_xor(s, m, 32);
    if (lane == 0) ysq[row] = s;
}

// ---------------------------------------------------------------------------
// Phase 0b: split X into bf16 hi/lo planes (row-major, same shape)
// ---------------------------------------------------------------------------
__global__ __launch_bounds__(256) void xcvt_kernel(const float* __restrict__ X,
                                                   __bf16* __restrict__ Xhi,
                                                   __bf16* __restrict__ Xlo) {
    const int i = blockIdx.x * 256 + threadIdx.x;  // grid covers NX*DDIM exactly
    float x = X[i];
    __bf16 h, l;
    split_bf16(x, h, l);
    Xhi[i] = h;
    Xlo[i] = l;
}

// ---------------------------------------------------------------------------
// Phase 1: per (M-block, Y-chunk): WMMA GEMM + running per-row top-8
// ---------------------------------------------------------------------------
__global__ __launch_bounds__(256) void knn_chunk_kernel(
    const __bf16* __restrict__ Xhi,
    const __bf16* __restrict__ Xlo,
    const float* __restrict__ Y,
    const float* __restrict__ ysq,
    float* __restrict__ pdist,
    int*   __restrict__ pidx) {

    __shared__ alignas(16) __bf16 sBhi[2][N_BLK * BPAD];
    __shared__ alignas(16) __bf16 sBlo[2][N_BLK * BPAD];
    __shared__ float sDist[M_BLK * DPAD];

    const int tid  = threadIdx.x;
    const int lane = tid & 31;
    const int w    = tid >> 5;              // wave 0..7, owns rows w*16..w*16+15
    const int ch   = blockIdx.x;            // Y chunk 0..31
    const int mb   = blockIdx.y;            // M block 0..63

    const int laneLo = lane & 15;
    const int selHi  = (lane >> 4) * 8;     // K sub-offset per ISA A/B layout

    // Cooperative-fill coordinates for this thread (2 float4 per slab)
    const int fr0 = (tid >> 3);             // rows 0..31  (+32 for second pass)
    const int fc  = (tid & 7) * 4;          // col group within 32-wide slab

    // Running top-8 (threads 0..127, one X row each) — kept in registers.
    float bd[KNEIGH];
    int   bi[KNEIGH];
#pragma unroll
    for (int k = 0; k < KNEIGH; ++k) { bd[k] = 3.0e38f; bi[k] = 0; }
    float worst = 3.0e38f;
    int   wpos  = 0;

    const int m = mb * M_BLK + w * 16 + laneLo;  // this lane's A row
    const __bf16* xhiRow = Xhi + (size_t)m * DDIM;
    const __bf16* xloRow = Xlo + (size_t)m * DDIM;

    for (int nt = 0; nt < NT_PER_CHUNK; ++nt) {
        const int n0 = ch * CHUNK_COLS + nt * N_BLK;  // first Y row of this tile

        v8f acc[4];
#pragma unroll
        for (int t = 0; t < 4; ++t) acc[t] = v8f{};

        // ---- prologue: fill ping buffer with slab k0 = 0 ----
#pragma unroll
        for (int it = 0; it < 2; ++it) {
            const int r = it * 32 + fr0;
            const float4 yv = *(const float4*)(Y + (size_t)(n0 + r) * DDIM + fc);
            const int b = r * BPAD + fc;
            const float e[4] = {yv.x, yv.y, yv.z, yv.w};
#pragma unroll
            for (int q = 0; q < 4; ++q)
                split_bf16(e[q], sBhi[0][b + q], sBlo[0][b + q]);
        }
        __syncthreads();

        // Cap unroll at 2: keeps ping-pong offsets immediate while bounding
        // live staging registers (full unroll pushed VGPRs past 256).
#pragma unroll 2
        for (int ks = 0; ks < K_STEPS; ++ks) {
            const int cur = ks & 1;
            const int k0  = ks * KT;
            const bool haveNext = (ks + 1) < K_STEPS;

            // Issue next slab's Y global loads early (hide behind WMMAs)
            float4 yn[2];
            if (haveNext) {
#pragma unroll
                for (int it = 0; it < 2; ++it) {
                    const int r = it * 32 + fr0;
                    yn[it] = *(const float4*)(Y + (size_t)(n0 + r) * DDIM +
                                              (k0 + KT) + fc);
                }
            }

            // A fragments from pre-split global planes (L0/L2 resident)
            FragB16 ah, al;
            {
                const __bf16* pa = xhiRow + k0 + selHi;
                ah.u[0] = *(const uint4*)pa;
                ah.u[1] = *(const uint4*)(pa + 16);
                const __bf16* pl = xloRow + k0 + selHi;
                al.u[0] = *(const uint4*)pl;
                al.u[1] = *(const uint4*)(pl + 16);
            }

#pragma unroll
            for (int t = 0; t < 4; ++t) {
                const int brow = t * 16 + laneLo;
                FragB16 bh, bl;
                const __bf16* pb = &sBhi[cur][brow * BPAD + selHi];
                bh.u[0] = *(const uint4*)pb;
                bh.u[1] = *(const uint4*)(pb + 16);
                const __bf16* pl = &sBlo[cur][brow * BPAD + selHi];
                bl.u[0] = *(const uint4*)pl;
                bl.u[1] = *(const uint4*)(pl + 16);

                acc[t] = __builtin_amdgcn_wmma_f32_16x16x32_bf16(
                    false, ah.v, false, bh.v, (short)0, acc[t], false, false);
                acc[t] = __builtin_amdgcn_wmma_f32_16x16x32_bf16(
                    false, ah.v, false, bl.v, (short)0, acc[t], false, false);
                acc[t] = __builtin_amdgcn_wmma_f32_16x16x32_bf16(
                    false, al.v, false, bh.v, (short)0, acc[t], false, false);
            }

            // Convert + store next slab into the pong buffer
            if (haveNext) {
#pragma unroll
                for (int it = 0; it < 2; ++it) {
                    const int r = it * 32 + fr0;
                    const int b = r * BPAD + fc;
                    const float e[4] = {yn[it].x, yn[it].y, yn[it].z, yn[it].w};
#pragma unroll
                    for (int q = 0; q < 4; ++q)
                        split_bf16(e[q], sBhi[1 - cur][b + q],
                                   sBlo[1 - cur][b + q]);
                }
            }
            __syncthreads();  // pong visible; ping reads complete before reuse
        }

        // distances -> LDS   (C layout: vgpr j, lanes0-15: M=j,N=lane; lanes16-31: M=j+8)
        const int rsel = (lane >> 4) * 8;
#pragma unroll
        for (int t = 0; t < 4; ++t) {
            const float ys = ysq[n0 + t * 16 + laneLo];
#pragma unroll
            for (int j = 0; j < 8; ++j) {
                const float d = ys - 2.0f * acc[t][j];
                sDist[(w * 16 + j + rsel) * DPAD + t * 16 + laneLo] = d;
            }
        }
        __syncthreads();

        // Running top-8 update: one thread per row scans 64 columns.
        if (tid < M_BLK) {
            const float* drow = &sDist[tid * DPAD];
            for (int c = 0; c < N_BLK; ++c) {
                const float d = drow[c];
                if (d < worst) {
#pragma unroll
                    for (int k = 0; k < KNEIGH; ++k)
                        if (k == wpos) { bd[k] = d; bi[k] = n0 + c; }
                    worst = bd[0]; wpos = 0;
#pragma unroll
                    for (int k = 1; k < KNEIGH; ++k)
                        if (bd[k] > worst) { worst = bd[k]; wpos = k; }
                }
            }
        }
        __syncthreads();  // scan done before dist tile reuse next iteration
    }

    if (tid < M_BLK) {
        const int row = mb * M_BLK + tid;
        const size_t off = ((size_t)row * N_CHUNKS + ch) * KNEIGH;
#pragma unroll
        for (int k = 0; k < KNEIGH; ++k) {
            pdist[off + k] = bd[k];
            pidx[off + k]  = bi[k];
        }
    }
}

// ---------------------------------------------------------------------------
// Phase 2: merge 32 partial top-8 lists per row, sort ascending, emit indices
// ---------------------------------------------------------------------------
__global__ __launch_bounds__(256) void knn_merge_kernel(
    const float* __restrict__ pdist, const int* __restrict__ pidx,
    int* __restrict__ out) {
    const int row = blockIdx.x * 256 + threadIdx.x;  // grid covers NX exactly
    const float* pd = pdist + (size_t)row * N_CHUNKS * KNEIGH;
    const int*   pi = pidx  + (size_t)row * N_CHUNKS * KNEIGH;

    float bd[KNEIGH];
    int   bi[KNEIGH];
#pragma unroll
    for (int k = 0; k < KNEIGH; ++k) { bd[k] = 3.0e38f; bi[k] = 0x7FFFFFFF; }
    float worst = 3.0e38f;
    int   wpos  = 0;

    for (int i = 0; i < N_CHUNKS * KNEIGH; ++i) {
        const float d  = pd[i];
        const int   id = pi[i];
        if (d < worst) {
#pragma unroll
            for (int k = 0; k < KNEIGH; ++k)
                if (k == wpos) { bd[k] = d; bi[k] = id; }
            worst = bd[0]; wpos = 0;
#pragma unroll
            for (int k = 1; k < KNEIGH; ++k)
                if (bd[k] > worst) { worst = bd[k]; wpos = k; }
        }
    }

    // sort ascending by (dist, index) — matches top_k ordering
#pragma unroll
    for (int i = 0; i < KNEIGH - 1; ++i) {
#pragma unroll
        for (int j = i + 1; j < KNEIGH; ++j) {
            bool doSwap = (bd[j] < bd[i]) || (bd[j] == bd[i] && bi[j] < bi[i]);
            float td = doSwap ? bd[i] : bd[j];
            int   ti = doSwap ? bi[i] : bi[j];
            bd[i] = doSwap ? bd[j] : bd[i];
            bi[i] = doSwap ? bi[j] : bi[i];
            bd[j] = td; bi[j] = ti;
        }
    }
#pragma unroll
    for (int k = 0; k < KNEIGH; ++k) out[(size_t)row * KNEIGH + k] = bi[k];
}

// ---------------------------------------------------------------------------
// Workspace layout (bytes):
//   [0)                      ysq     : NY*4            =   256 KB
//   [256KB)                  Xhi     : NX*DDIM*2       =     8 MB
//   [256KB+8MB)              Xlo     : NX*DDIM*2       =     8 MB
//   [256KB+16MB)             pdist   : NX*32*8*4       =     8 MB
//   [256KB+24MB)             pidx    : NX*32*8*4       =     8 MB
//   total ~= 33.8 MB
// ---------------------------------------------------------------------------
extern "C" void kernel_launch(void* const* d_in, const int* in_sizes, int n_in,
                              void* d_out, int out_size, void* d_ws, size_t ws_size,
                              hipStream_t stream) {
    (void)in_sizes; (void)n_in; (void)out_size; (void)ws_size;
    const float* X = (const float*)d_in[0];
    const float* Y = (const float*)d_in[1];

    char* ws = (char*)d_ws;
    float*  ysq   = (float*)ws;
    __bf16* Xhi   = (__bf16*)(ws + (256u << 10));
    __bf16* Xlo   = (__bf16*)(ws + (256u << 10) + (size_t)NX * DDIM * 2);
    float*  pdist = (float*)(ws + (256u << 10) + (size_t)NX * DDIM * 4);
    int*    pidx  = (int*)((char*)pdist + (size_t)NX * N_CHUNKS * KNEIGH * 4);

    ysq_kernel<<<NY / 8, 256, 0, stream>>>(Y, ysq);
    xcvt_kernel<<<(NX * DDIM) / 256, 256, 0, stream>>>(X, Xhi, Xlo);

    dim3 g1(N_CHUNKS, NX / M_BLK);
    knn_chunk_kernel<<<g1, 256, 0, stream>>>(Xhi, Xlo, Y, ysq, pdist, pidx);

    knn_merge_kernel<<<NX / 256, 256, 0, stream>>>(pdist, pidx, (int*)d_out);
}
